// GRUGAT_82076825026992
// MI455X (gfx1250) — matrix operand order repaired and checked by reading, not compiled, not run on hardware
//
#include <hip/hip_runtime.h>
#include <hip/hip_bf16.h>
#include <math.h>
#include <stdint.h>

#define B_   8
#define NODE_ 1024
#define T_   12
#define DIN_ 16
#define HID_ 32
#define GOUT_ 32
#define PLEN_ 6
#define HEADS_ 3
#define FIN_ 384          // HID*T
#define C_   384          // GOUT*T
#define N_   8192         // B*NODE
#define E_   65536
#define HCOL_ 1152        // HEADS*C

typedef float v8f  __attribute__((ext_vector_type(8)));
typedef float v2f  __attribute__((ext_vector_type(2)));
typedef __bf16 bf16x8  __attribute__((ext_vector_type(8)));
typedef __bf16 bf16x16 __attribute__((ext_vector_type(16)));
typedef int v4i __attribute__((ext_vector_type(4)));
typedef int v8i __attribute__((ext_vector_type(8)));

__device__ __forceinline__ __bf16 f2bf(float f) {
    unsigned u = __float_as_uint(f);
    unsigned r = (u + 0x7FFFu + ((u >> 16) & 1u)) >> 16;
    unsigned short s = (unsigned short)r;
    __bf16 b;
    __builtin_memcpy(&b, &s, 2);
    return b;
}

__device__ __forceinline__ void atomicMaxF(float* addr, float v) {
    if (v >= 0.0f) atomicMax((int*)addr, __float_as_int(v));
    else           atomicMin((unsigned int*)addr, __float_as_uint(v));
}

// LDS byte offset of a generic pointer to __shared__ (aperture in high word,
// offset in addr[31:0] per CDNA5 ISA 10.2).
__device__ __forceinline__ unsigned lds_offset(const void* p) {
    return (unsigned)(uintptr_t)p;
}

// ---------------------------------------------------------------- f32 -> bf16
__global__ void k_f2bf(const float* __restrict__ src, __bf16* __restrict__ dst, int n) {
    int i = blockIdx.x * 256 + threadIdx.x;
    if (i < n) dst[i] = f2bf(src[i]);
}

// ---------------------------------------------------------------- init scratch
__global__ void k_init(float* __restrict__ accum, float* __restrict__ denom,
                       float* __restrict__ amax, float* __restrict__ pooled) {
    int i = blockIdx.x * 256 + threadIdx.x;
    for (int j = i; j < N_ * C_; j += gridDim.x * 256) accum[j] = 0.0f;
    if (i < N_ * HEADS_) { denom[i] = 0.0f; amax[i] = -3.0e38f; }
    if (i < B_ * C_) pooled[i] = -3.0e38f;
}

// ---------------------------------------------------------------- edge coefficient
__global__ void k_edge_coef(const float* __restrict__ Wedge,
                            const float* __restrict__ att_edge,
                            float* __restrict__ ce) {
    int hd = threadIdx.x >> 5, ln = threadIdx.x & 31;   // blockDim = 96
    float s = 0.0f;
    for (int c = ln; c < C_; c += 32) s += Wedge[hd * C_ + c] * att_edge[hd * C_ + c];
    for (int m = 16; m; m >>= 1) s += __shfl_xor(s, m, 32);
    if (ln == 0) ce[hd] = s;
}

// ---------------------------------------------------------------- GRU (fp32 WMMA 16x16x4)
// One wave handles 16 rows; x tile staged with GLOBAL_LOAD_ASYNC_TO_LDS_B128
// (ASYNCcnt path); gi/gh via V_WMMA_F32_16X16X4_F32.
__global__ __launch_bounds__(256) void k_gru(const float* __restrict__ x,
                                             const float* __restrict__ Wih,
                                             const float* __restrict__ Whh,
                                             const float* __restrict__ bih,
                                             const float* __restrict__ bhh,
                                             __bf16* __restrict__ Abf) {
    __shared__ float sWih[96 * 16];
    __shared__ float sWhh[96 * 32];
    __shared__ float sb[192];
    __shared__ __align__(16) float sX[8][16 * 16];
    __shared__ float sH[8][16 * 32];

    const int tid = threadIdx.x;
    for (int i = tid; i < 96 * 16; i += 256) sWih[i] = Wih[i];
    for (int i = tid; i < 96 * 32; i += 256) sWhh[i] = Whh[i];
    if (tid < 96) sb[tid] = bih[tid];
    else if (tid < 192) sb[tid] = bhh[tid - 96];

    const int wv = tid >> 5, ln = tid & 31;
    const int m0 = (blockIdx.x * 8 + wv) * 16;
    const int hi = (ln >= 16) ? 1 : 0;
    const int lc = ln & 15;
    const int klo = hi ? 2 : 0;

    for (int i = ln; i < 512; i += 32) sH[wv][i] = 0.0f;
    __syncthreads();

    v8f hfrag[2];
    #pragma unroll
    for (int j = 0; j < 2; ++j)
        #pragma unroll
        for (int i = 0; i < 8; ++i) hfrag[j][i] = 0.0f;

    const unsigned ldsX = lds_offset(&sX[wv][0]);

    for (int t = 0; t < T_; ++t) {
        // async-stage xt tile (16x16 f32 = 64 x b128) into LDS
        #pragma unroll
        for (int i = 0; i < 2; ++i) {
            int idx = i * 32 + ln;                 // 0..63
            int r = idx >> 2, c4 = (idx & 3) * 4;
            unsigned ldsoff = ldsX + (unsigned)(r * 16 + c4) * 4u;
            unsigned long long ga = (unsigned long long)(uintptr_t)
                &x[(size_t)(m0 + r) * (T_ * DIN_) + t * DIN_ + c4];
            asm volatile("global_load_async_to_lds_b128 %0, %1, off"
                         :: "v"(ldsoff), "v"(ga) : "memory");
        }
        asm volatile("s_wait_asynccnt 0x0" ::: "memory");
        __syncthreads();

        v8f gi[6], gh[6];
        #pragma unroll
        for (int nt = 0; nt < 6; ++nt) {
            v8f acc;
            #pragma unroll
            for (int i = 0; i < 8; ++i) acc[i] = 0.0f;
            #pragma unroll
            for (int kc = 0; kc < 16; kc += 4) {
                v2f a = *(const v2f*)&sX[wv][lc * 16 + kc + klo];
                v2f b = *(const v2f*)&sWih[(nt * 16 + lc) * 16 + kc + klo];
                acc = __builtin_amdgcn_wmma_f32_16x16x4_f32(false, a, false, b,
                                                            (short)0, acc, false, false);
            }
            gi[nt] = acc;
        }
        #pragma unroll
        for (int nt = 0; nt < 6; ++nt) {
            v8f acc;
            #pragma unroll
            for (int i = 0; i < 8; ++i) acc[i] = 0.0f;
            #pragma unroll
            for (int kc = 0; kc < 32; kc += 4) {
                v2f a = *(const v2f*)&sH[wv][lc * 32 + kc + klo];
                v2f b = *(const v2f*)&sWhh[(nt * 16 + lc) * 32 + kc + klo];
                acc = __builtin_amdgcn_wmma_f32_16x16x4_f32(false, a, false, b,
                                                            (short)0, acc, false, false);
            }
            gh[nt] = acc;
        }
        __syncthreads();   // done reading sH/sX; about to overwrite sH

        #pragma unroll
        for (int ht = 0; ht < 2; ++ht) {
            int col = ht * 16 + lc;
            float br_i = sb[col],        bz_i = sb[32 + col],  bn_i = sb[64 + col];
            float br_h = sb[96 + col],   bz_h = sb[128 + col], bn_h = sb[160 + col];
            v8f hnew;
            #pragma unroll
            for (int i = 0; i < 8; ++i) {
                float gr = gi[ht][i] + br_i + gh[ht][i] + br_h;
                float gz = gi[2 + ht][i] + bz_i + gh[2 + ht][i] + bz_h;
                float gn_i = gi[4 + ht][i] + bn_i;
                float gn_h = gh[4 + ht][i] + bn_h;
                float r = 1.0f / (1.0f + __expf(-gr));
                float z = 1.0f / (1.0f + __expf(-gz));
                float nn = tanhf(gn_i + r * gn_h);
                hnew[i] = (1.0f - z) * nn + z * hfrag[ht][i];
            }
            hfrag[ht] = hnew;
            #pragma unroll
            for (int i = 0; i < 8; ++i) {
                int row = i + (hi ? 8 : 0);
                sH[wv][row * 32 + col] = hnew[i];
                Abf[(size_t)(m0 + row) * FIN_ + t * HID_ + col] = f2bf(hnew[i]);
            }
        }
        __syncthreads();
    }
}

// ---------------------------------------------------------------- GAT GEMM (bf16 WMMA 16x16x32)
// Block owns one 64-column group; the 64x384 bf16 W tile (48KB, contiguous)
// is DMA'd into LDS with a single TENSOR_LOAD_TO_LDS (TDM / TENSORcnt path).
// Each of 8 waves computes a 16x64 C-tile; B fragments come from LDS.
__global__ __launch_bounds__(256) void k_gemm(const __bf16* __restrict__ A,
                                              const __bf16* __restrict__ W,
                                              float* __restrict__ H) {
    __shared__ __align__(32) __bf16 sW[64 * FIN_];          // 49152 B
    const int tid = threadIdx.x, wv = tid >> 5, ln = tid & 31;
    const int ng = blockIdx.x % 18;
    const int m0 = ((blockIdx.x / 18) * 8 + wv) * 16;
    const int n0 = ng * 64;
    const int lc = ln & 15;
    const int hi = (ln >= 16) ? 1 : 0;

    if (tid < 32) {            // wave 0 issues the TDM load (EXEC ignored by TDM)
        unsigned long long ga = (unsigned long long)(uintptr_t)(W + (size_t)n0 * FIN_);
        v4i g0;
        g0.x = 1;                                               // count=1, user desc
        g0.y = (int)lds_offset(&sW[0]);                         // lds_addr
        g0.z = (int)(unsigned)(ga & 0xFFFFFFFFull);             // global_addr[31:0]
        g0.w = (int)(((unsigned)(ga >> 32) & 0x01FFFFFFu)       // global_addr[56:32]
                     | 0x80000000u);                            // type=2 (bits 127:126)
        v8i g1;
        g1[0] = (1 << 16);                    // workgroup_mask=0, data_size=1 (2B)
        g1[1] = (FIN_ & 0xFFFF) << 16;        // tensor_dim0[15:0] = 384
        g1[2] = (64 << 16);                   // tensor_dim0[31:16]=0 | tensor_dim1=64
        g1[3] = (FIN_ << 16);                 // tensor_dim1 hi=0 | tile_dim0=384
        g1[4] = 64;                           // tile_dim1=64, tile_dim2=0
        g1[5] = FIN_;                         // tensor_dim0_stride[31:0] = 384
        g1[6] = 0;                            // stride hi / dim1_stride lo (unused, 2D)
        g1[7] = 0;
        asm volatile("tensor_load_to_lds %0, %1" :: "s"(g0), "s"(g1) : "memory");
        __builtin_amdgcn_s_wait_tensorcnt(0);
    }
    __syncthreads();

    v8f acc[4];
    #pragma unroll
    for (int j = 0; j < 4; ++j)
        #pragma unroll
        for (int i = 0; i < 8; ++i) acc[j][i] = 0.0f;

    for (int kc = 0; kc < FIN_; kc += 32) {
        // A fragment: lanes 0-15 carry K lo half, 16-31 K hi half (contiguous, matches B)
        const __bf16* ap = A + (size_t)(m0 + lc) * FIN_ + kc + (hi ? 16 : 0);
        bf16x8 alo = *(const bf16x8*)ap;
        bf16x8 ahi = *(const bf16x8*)(ap + 8);
        bf16x16 af;
        #pragma unroll
        for (int i = 0; i < 8; ++i) { af[i] = alo[i]; af[8 + i] = ahi[i]; }
        if (kc + 32 < FIN_) __builtin_prefetch(ap + 32, 0, 1);

        #pragma unroll
        for (int j = 0; j < 4; ++j) {
            // B fragment from LDS: lanes 0-15 carry K lo half, 16-31 K hi half
            bf16x16 bfrag = *(const bf16x16*)&sW[(j * 16 + lc) * FIN_ + kc + (hi ? 16 : 0)];
            acc[j] = __builtin_amdgcn_wmma_f32_16x16x32_bf16(false, af, false, bfrag,
                                                             (short)0, acc[j], false, false);
        }
    }
    #pragma unroll
    for (int j = 0; j < 4; ++j)
        #pragma unroll
        for (int i = 0; i < 8; ++i)
            H[(size_t)(m0 + i + (hi ? 8 : 0)) * HCOL_ + n0 + j * 16 + lc] = acc[j][i];
}

// ---------------------------------------------------------------- per-node attention logits
__global__ __launch_bounds__(256) void k_attn(const float* __restrict__ H,
                                              const float* __restrict__ att_src,
                                              const float* __restrict__ att_dst,
                                              float* __restrict__ a_src,
                                              float* __restrict__ a_dst) {
    int wv = threadIdx.x >> 5, ln = threadIdx.x & 31;
    int gw = blockIdx.x * 8 + wv;                 // N_*HEADS_ = 24576
    int n = gw / HEADS_, hd = gw % HEADS_;
    const float* hp = H + (size_t)n * HCOL_ + hd * C_;
    float s = 0.0f, d = 0.0f;
    for (int c = ln; c < C_; c += 32) {
        float hv = hp[c];
        s += hv * att_src[hd * C_ + c];
        d += hv * att_dst[hd * C_ + c];
    }
    for (int m = 16; m; m >>= 1) { s += __shfl_xor(s, m, 32); d += __shfl_xor(d, m, 32); }
    if (ln == 0) { a_src[n * HEADS_ + hd] = s; a_dst[n * HEADS_ + hd] = d; }
}

// ---------------------------------------------------------------- edge pass 1: logits + segment max
__global__ void k_edge_a(const int* __restrict__ ei, const float* __restrict__ eattr,
                         const float* __restrict__ ce,
                         const float* __restrict__ a_src, const float* __restrict__ a_dst,
                         float* __restrict__ aw, float* __restrict__ amax) {
    int gid = blockIdx.x * 256 + threadIdx.x;     // E_*HEADS_
    int e = gid / HEADS_, hd = gid % HEADS_;
    int s = ei[e], d = ei[E_ + e];
    float a = a_src[s * HEADS_ + hd] + a_dst[d * HEADS_ + hd] + eattr[e] * ce[hd];
    a = (a > 0.0f) ? a : 0.2f * a;                // leaky_relu 0.2
    aw[gid] = a;
    atomicMaxF(&amax[d * HEADS_ + hd], a);
}

// ---------------------------------------------------------------- edge pass 2: exp + segment sum
__global__ void k_edge_b(const int* __restrict__ ei, float* __restrict__ aw,
                         const float* __restrict__ amax, float* __restrict__ denom) {
    int gid = blockIdx.x * 256 + threadIdx.x;
    int e = gid / HEADS_, hd = gid % HEADS_;
    int d = ei[E_ + e];
    float t = __expf(aw[gid] - amax[d * HEADS_ + hd]);
    aw[gid] = t;
    atomicAdd(&denom[d * HEADS_ + hd], t);
}

// ---------------------------------------------------------------- edge pass 3: message scatter
// One wave per edge; head-mean folded in.  h[src] gathers are L2-resident
// (H = 37.7MB < 192MB L2), so this phase is L2-bandwidth bound, not HBM.
__global__ __launch_bounds__(256) void k_edge_c(const int* __restrict__ ei,
                                                const float* __restrict__ aw,
                                                const float* __restrict__ denom,
                                                const float* __restrict__ H,
                                                float* __restrict__ accum) {
    int wv = threadIdx.x >> 5, ln = threadIdx.x & 31;
    int e = blockIdx.x * 8 + wv;                  // E_
    int s = ei[e], d = ei[E_ + e];
    float w0 = aw[e * 3 + 0] / (denom[d * 3 + 0] + 1e-16f);
    float w1 = aw[e * 3 + 1] / (denom[d * 3 + 1] + 1e-16f);
    float w2 = aw[e * 3 + 2] / (denom[d * 3 + 2] + 1e-16f);
    const float third = 1.0f / 3.0f;
    const float* hp = H + (size_t)s * HCOL_;
    float* ap = accum + (size_t)d * C_;
    for (int c = ln; c < C_; c += 32) {
        float v = (w0 * hp[c] + w1 * hp[C_ + c] + w2 * hp[2 * C_ + c]) * third;
        atomicAdd(&ap[c], v);
    }
}

// ---------------------------------------------------------------- node finalize + predictor + pooled max
__global__ void k_node(const float* __restrict__ accum, const float* __restrict__ gat_bias,
                       const int* __restrict__ batch_vec,
                       const float* __restrict__ Wp1, const float* __restrict__ bp1,
                       const float* __restrict__ Wp3, const float* __restrict__ bp3,
                       float* __restrict__ pooled, float* __restrict__ out) {
    int n = blockIdx.x * 256 + threadIdx.x;       // N_
    int b = batch_vec[n];
    float p[T_];
    #pragma unroll
    for (int t = 0; t < T_; ++t) {
        float acc = bp1[0];
        #pragma unroll
        for (int cc = 0; cc < GOUT_; ++cc) {
            int col = t * GOUT_ + cc;
            float g = accum[(size_t)n * C_ + col] + gat_bias[col];
            atomicMaxF(&pooled[b * C_ + col], g);
            acc += g * Wp1[cc];
        }
        p[t] = fmaxf(acc, 0.0f);
    }
    #pragma unroll
    for (int k = 0; k < PLEN_; ++k) {
        float s = bp3[k];
        #pragma unroll
        for (int t = 0; t < T_; ++t) s += p[t] * Wp3[k * T_ + t];
        out[n * PLEN_ + k] = fmaxf(s, 0.0f);
    }
}

// ---------------------------------------------------------------- classifier + softmax
__global__ __launch_bounds__(256) void k_cls(const float* __restrict__ pooled,
                                             const float* __restrict__ Wc1, const float* __restrict__ bc1,
                                             const float* __restrict__ Wc2, const float* __restrict__ bc2,
                                             float* __restrict__ out) {
    __shared__ float l1s[B_][GOUT_];
    __shared__ float l2s[B_][2];
    int tid = threadIdx.x;
    int b = tid >> 5, o = tid & 31;
    float s = bc1[o];
    for (int c = 0; c < C_; ++c) s += pooled[b * C_ + c] * Wc1[o * C_ + c];
    l1s[b][o] = s;
    __syncthreads();
    if (tid < 16) {
        int bb = tid >> 1, j = tid & 1;
        float v = bc2[j];
        #pragma unroll
        for (int k = 0; k < GOUT_; ++k) v += l1s[bb][k] * Wc2[j * GOUT_ + k];
        l2s[bb][j] = v;
    }
    __syncthreads();
    if (tid < 16) {
        int bb = tid >> 1, j = tid & 1;
        float m = fmaxf(l2s[bb][0], l2s[bb][1]);
        float e0 = __expf(l2s[bb][0] - m), e1 = __expf(l2s[bb][1] - m);
        out[N_ * PLEN_ + bb * 2 + j] = (j ? e1 : e0) / (e0 + e1);
    }
}

// ================================================================ launch
extern "C" void kernel_launch(void* const* d_in, const int* in_sizes, int n_in,
                              void* d_out, int out_size, void* d_ws, size_t ws_size,
                              hipStream_t stream) {
    (void)in_sizes; (void)n_in; (void)out_size; (void)ws_size;
    const float* x        = (const float*)d_in[0];
    const int*   ei       = (const int*)  d_in[1];
    const float* eattr    = (const float*)d_in[2];
    const int*   batchv   = (const int*)  d_in[3];
    /* alpha = d_in[4], identity in forward */
    const float* Wih      = (const float*)d_in[5];
    const float* Whh      = (const float*)d_in[6];
    const float* bih      = (const float*)d_in[7];
    const float* bhh      = (const float*)d_in[8];
    const float* Wgat     = (const float*)d_in[9];
    const float* att_src  = (const float*)d_in[10];
    const float* att_dst  = (const float*)d_in[11];
    const float* Wedge    = (const float*)d_in[12];
    const float* att_edge = (const float*)d_in[13];
    const float* gat_bias = (const float*)d_in[14];
    const float* Wp1      = (const float*)d_in[15];
    const float* bp1      = (const float*)d_in[16];
    const float* Wp3      = (const float*)d_in[17];
    const float* bp3      = (const float*)d_in[18];
    const float* Wc1      = (const float*)d_in[19];
    const float* bc1      = (const float*)d_in[20];
    const float* Wc2      = (const float*)d_in[21];
    const float* bc2      = (const float*)d_in[22];
    float* out = (float*)d_out;

    char* ws = (char*)d_ws;
    __bf16* Abf   = (__bf16*)(ws);                       //  6,291,456 B
    __bf16* Wbf   = (__bf16*)(ws + 6291456);             //    884,736 B
    float*  H     = (float*) (ws + 7176192);             // 37,748,736 B
    float*  a_src = (float*) (ws + 44924928);            //     98,304 B
    float*  a_dst = (float*) (ws + 45023232);            //     98,304 B
    float*  amax  = (float*) (ws + 45121536);            //     98,304 B
    float*  denom = (float*) (ws + 45219840);            //     98,304 B
    float*  aw    = (float*) (ws + 45318144);            //    786,432 B
    float*  accum = (float*) (ws + 46104576);            // 12,582,912 B
    float*  pooled= (float*) (ws + 58687488);            //     12,288 B
    float*  ce    = (float*) (ws + 58699776);            //         12 B

    k_f2bf     <<<1728, 256, 0, stream>>>(Wgat, Wbf, HCOL_ * FIN_);
    k_init     <<<12288, 256, 0, stream>>>(accum, denom, amax, pooled);
    k_edge_coef<<<1, 96, 0, stream>>>(Wedge, att_edge, ce);
    k_gru      <<<64, 256, 0, stream>>>(x, Wih, Whh, bih, bhh, Abf);
    k_gemm     <<<1152, 256, 0, stream>>>(Abf, Wbf, H);
    k_attn     <<<3072, 256, 0, stream>>>(H, att_src, att_dst, a_src, a_dst);
    k_edge_a   <<<768, 256, 0, stream>>>(ei, eattr, ce, a_src, a_dst, aw, amax);
    k_edge_b   <<<768, 256, 0, stream>>>(ei, aw, amax, denom);
    k_edge_c   <<<8192, 256, 0, stream>>>(ei, aw, denom, H, accum);
    k_node     <<<32, 256, 0, stream>>>(accum, gat_bias, batchv, Wp1, bp1, Wp3, bp3, pooled, out);
    k_cls      <<<1, 256, 0, stream>>>(pooled, Wc1, bc1, Wc2, bc2, out);
}
